// GRUCoordinateHead_30511447671058
// MI455X (gfx1250) — matrix operand order repaired and checked by reading, not compiled, not run on hardware
//
#include <hip/hip_runtime.h>
#include <hip/hip_bf16.h>
#include <math.h>

// ---------------------------------------------------------------------------
// GRU coordinate head, fused recurrent kernel for gfx1250 (MI455X).
// B=1024, T=256, C=512, H=256.  bf16 WMMA (16x16x32) with f32 accumulation.
// ---------------------------------------------------------------------------

#define Bsz  1024
#define Tlen 256
#define CF   512
#define Hdim 256
#define G3H  768          // 3*H
#define CIN  (CF + 2)     // 514
#define BM   16           // batch rows per block (one M tile)
#define NT   256          // threads per block = 8 waves

typedef __bf16 bf16_t;
typedef __attribute__((ext_vector_type(16))) __bf16 v16bf;
typedef __attribute__((ext_vector_type(8)))  __bf16 v8bf;
typedef __attribute__((ext_vector_type(8)))  float  v8f;

// --- fragment loaders -------------------------------------------------------
// A fragment (16x32 bf16): lane<16 -> row=lane, K = k0..k0+15
//                          lane>=16 -> row=lane-16, K = k0+16..k0+31.
__device__ __forceinline__ v16bf load_a_frag(const bf16_t* p) {
    v8bf lo = *(const v8bf*)(p);
    v8bf hi = *(const v8bf*)(p + 8);
    return __builtin_shufflevector(lo, hi, 0,1,2,3,4,5,6,7,8,9,10,11,12,13,14,15);
}
// B fragment (32x16 bf16): lane holds column n=lane&15, 16 contiguous K starting at
// k0 + (lane>>4)*16.  Since B[k,n] = W[n,k] (row-major W), this is contiguous.
__device__ __forceinline__ v16bf load_b_frag(const bf16_t* p) {
    v8bf lo = *(const v8bf*)(p);
    v8bf hi = *(const v8bf*)(p + 8);
    return __builtin_shufflevector(lo, hi, 0,1,2,3,4,5,6,7,8,9,10,11,12,13,14,15);
}

__device__ __forceinline__ v8f wmma_bf16(v16bf a, v16bf b, v8f c) {
    return __builtin_amdgcn_wmma_f32_16x16x32_bf16(
        /*neg_a=*/false, a, /*neg_b=*/false, b,
        /*c_mod=*/(short)0, c, /*reuse_a=*/false, /*reuse_b=*/false);
}

__device__ __forceinline__ float sigmoid_f(float x) {
    return 1.0f / (1.0f + __expf(-x));
}

// --- weight prep: f32 -> bf16, drop the 2 coord columns of W_ih -------------
__global__ __launch_bounds__(256)
void gru_prep_weights(const float* __restrict__ W_ih, const float* __restrict__ W_hh,
                      bf16_t* __restrict__ wih_bf, bf16_t* __restrict__ whh_bf) {
    int idx = blockIdx.x * blockDim.x + threadIdx.x;
    const int n1 = G3H * CF;
    if (idx < n1) {
        int r = idx / CF, c = idx - r * CF;
        wih_bf[idx] = (bf16_t)W_ih[r * CIN + c];
    } else {
        int j = idx - n1;
        if (j < G3H * Hdim) whh_bf[j] = (bf16_t)W_hh[j];
    }
}

// --- fused recurrent kernel -------------------------------------------------
__global__ __launch_bounds__(NT)
void gru_coord_head_kernel(const float* __restrict__ features,
                           const float* __restrict__ last_coord,
                           const float* __restrict__ W_ih,   // f32, for coord cols
                           const float* __restrict__ b_ih,
                           const float* __restrict__ b_hh,
                           const float* __restrict__ W_proj,
                           const float* __restrict__ b_proj,
                           const bf16_t* __restrict__ wih_bf,  // (768,512) bf16
                           const bf16_t* __restrict__ whh_bf,  // (768,256) bf16
                           float* __restrict__ out) {
    __shared__ bf16_t sfeat[2][BM * CF];   // double-buffered feature tile (bf16)
    __shared__ float  sh_h [BM * Hdim];    // h (f32) for gate math / projection
    __shared__ bf16_t sh_hb[BM * Hdim];    // h (bf16) as WMMA A operand
    __shared__ float  scoord[BM][2];

    const int tid   = threadIdx.x;
    const int lane  = tid & 31;
    const int wave  = tid >> 5;            // 0..7
    const int lr    = lane & 15;
    const int lhalf = lane >> 4;
    const int b0    = blockIdx.x * BM;

    // ---- init h = 0, load coords ----
    for (int i = tid; i < BM * Hdim; i += NT) { sh_h[i] = 0.0f; sh_hb[i] = (bf16_t)0.0f; }
    if (tid < BM * 2) scoord[tid >> 1][tid & 1] = last_coord[(b0 + (tid >> 1)) * 2 + (tid & 1)];

    // ---- per-lane gate constants (hoisted out of the time loop) ----
    float biasr[2], biasz[2], bni[2], bnh[2];
    float wr0[2], wr1[2], wz0[2], wz1[2], wn0[2], wn1[2];
#pragma unroll
    for (int s = 0; s < 2; ++s) {
        const int hc = wave + s * 8;
        const int Nr = 0 * Hdim + hc * 16 + lr;
        const int Nz = 1 * Hdim + hc * 16 + lr;
        const int Nn = 2 * Hdim + hc * 16 + lr;
        biasr[s] = b_ih[Nr] + b_hh[Nr];
        biasz[s] = b_ih[Nz] + b_hh[Nz];
        bni[s]   = b_ih[Nn];
        bnh[s]   = b_hh[Nn];
        wr0[s] = W_ih[Nr * CIN + CF];  wr1[s] = W_ih[Nr * CIN + CF + 1];
        wz0[s] = W_ih[Nz * CIN + CF];  wz1[s] = W_ih[Nz * CIN + CF + 1];
        wn0[s] = W_ih[Nn * CIN + CF];  wn1[s] = W_ih[Nn * CIN + CF + 1];
    }

    // ---- stage features for t=0 ----
    const int sm = tid >> 4;           // 0..15 : batch row in tile
    const int sc = (tid & 15) * 32;    // 32 channels per thread
    {
        const float* src = features + ((size_t)(b0 + sm) * Tlen + 0) * CF + sc;
#pragma unroll
        for (int q = 0; q < 8; ++q) {
            float4 v = ((const float4*)src)[q];
            bf16_t* dst = &sfeat[0][sm * CF + sc + q * 4];
            dst[0] = (bf16_t)v.x; dst[1] = (bf16_t)v.y;
            dst[2] = (bf16_t)v.z; dst[3] = (bf16_t)v.w;
        }
    }
    __syncthreads();

    for (int t = 0; t < Tlen; ++t) {
        const int cur = t & 1;

        // prefetch features for t+1 into registers (hidden behind WMMA)
        float4 pf[8];
        const bool havePf = (t + 1 < Tlen);
        if (havePf) {
            const float* src = features + ((size_t)(b0 + sm) * Tlen + (t + 1)) * CF + sc;
#pragma unroll
            for (int q = 0; q < 8; ++q) pf[q] = ((const float4*)src)[q];
        }

        // accumulators: [hc-slot][ gi_r, gi_z, gi_n, gh_r, gh_z, gh_n ]
        v8f acc[2][6];
#pragma unroll
        for (int s = 0; s < 2; ++s)
#pragma unroll
            for (int g = 0; g < 6; ++g)
                acc[s][g] = (v8f){0.f,0.f,0.f,0.f,0.f,0.f,0.f,0.f};

#pragma unroll
        for (int s = 0; s < 2; ++s) {
            const int hc = wave + s * 8;
            // gi (feature part): K = 512 -> 16 k-steps of 32
#pragma unroll 4
            for (int ks = 0; ks < CF / 32; ++ks) {
                const int k0 = ks * 32;
                v16bf a = load_a_frag(&sfeat[cur][lr * CF + k0 + lhalf * 16]);
#pragma unroll
                for (int g = 0; g < 3; ++g) {
                    const int row = g * Hdim + hc * 16 + lr;
                    v16bf b = load_b_frag(&wih_bf[(size_t)row * CF + k0 + lhalf * 16]);
                    acc[s][g] = wmma_bf16(a, b, acc[s][g]);
                }
            }
            // gh (recurrent part): K = 256 -> 8 k-steps of 32
#pragma unroll 4
            for (int ks = 0; ks < Hdim / 32; ++ks) {
                const int k0 = ks * 32;
                v16bf a = load_a_frag(&sh_hb[lr * Hdim + k0 + lhalf * 16]);
#pragma unroll
                for (int g = 0; g < 3; ++g) {
                    const int row = g * Hdim + hc * 16 + lr;
                    v16bf b = load_b_frag(&whh_bf[(size_t)row * Hdim + k0 + lhalf * 16]);
                    acc[s][3 + g] = wmma_bf16(a, b, acc[s][3 + g]);
                }
            }
        }

        // store prefetched features into the other buffer (read next iter)
        if (havePf) {
            bf16_t* dst = &sfeat[cur ^ 1][sm * CF + sc];
#pragma unroll
            for (int q = 0; q < 8; ++q) {
                dst[q * 4 + 0] = (bf16_t)pf[q].x; dst[q * 4 + 1] = (bf16_t)pf[q].y;
                dst[q * 4 + 2] = (bf16_t)pf[q].z; dst[q * 4 + 3] = (bf16_t)pf[q].w;
            }
        }

        __syncthreads();   // all waves done reading sh_hb for this step

        // ---- gate math in accumulator registers, write h_new ----
#pragma unroll
        for (int s = 0; s < 2; ++s) {
            const int hc = wave + s * 8;
#pragma unroll
            for (int v = 0; v < 8; ++v) {
                const int m  = v + lhalf * 8;          // C/D layout (ISA 7.12.2)
                const float c0 = scoord[m][0];
                const float c1 = scoord[m][1];
                const float r = sigmoid_f(acc[s][0][v] + acc[s][3][v] + biasr[s]
                                          + c0 * wr0[s] + c1 * wr1[s]);
                const float z = sigmoid_f(acc[s][1][v] + acc[s][4][v] + biasz[s]
                                          + c0 * wz0[s] + c1 * wz1[s]);
                const float i_n = acc[s][2][v] + bni[s] + c0 * wn0[s] + c1 * wn1[s];
                const float h_n = acc[s][5][v] + bnh[s];
                const float nn  = tanhf(i_n + r * h_n);
                const int idx = m * Hdim + hc * 16 + lr;
                const float hold = sh_h[idx];
                const float hnew = (1.0f - z) * nn + z * hold;
                sh_h[idx]  = hnew;
                sh_hb[idx] = (bf16_t)hnew;
            }
        }
        __syncthreads();   // h_new visible for projection

        // ---- projection: disp = h_new @ W_proj^T + b_proj (16 lanes per row) ----
        {
            const int l = tid & 15;
            float a0 = 0.0f, a1 = 0.0f;
            for (int k = l; k < Hdim; k += 16) {
                const float hv = sh_h[sm * Hdim + k];
                a0 += hv * W_proj[k];
                a1 += hv * W_proj[Hdim + k];
            }
#pragma unroll
            for (int off = 8; off > 0; off >>= 1) {
                a0 += __shfl_xor(a0, off);
                a1 += __shfl_xor(a1, off);
            }
            if (l == 0) {
                const float d0 = a0 + b_proj[0];
                const float d1 = a1 + b_proj[1];
                const size_t o = ((size_t)(b0 + sm) * Tlen + t) * 2;
                out[o + 0] = d0;
                out[o + 1] = d1;
                scoord[sm][0] += d0;
                scoord[sm][1] += d1;
            }
        }
        __syncthreads();   // coord + staged features ready for t+1
    }
}

// ---------------------------------------------------------------------------
extern "C" void kernel_launch(void* const* d_in, const int* in_sizes, int n_in,
                              void* d_out, int out_size, void* d_ws, size_t ws_size,
                              hipStream_t stream) {
    (void)in_sizes; (void)n_in; (void)out_size; (void)ws_size;
    const float* features   = (const float*)d_in[0];
    const float* last_coord = (const float*)d_in[1];
    const float* W_ih       = (const float*)d_in[2];
    const float* W_hh       = (const float*)d_in[3];
    const float* b_ih       = (const float*)d_in[4];
    const float* b_hh       = (const float*)d_in[5];
    const float* W_proj     = (const float*)d_in[6];
    const float* b_proj     = (const float*)d_in[7];

    bf16_t* wih_bf = (bf16_t*)d_ws;                                  // 768*512 bf16
    bf16_t* whh_bf = (bf16_t*)((char*)d_ws + (size_t)G3H * CF * 2);  // 768*256 bf16

    const int nconv = G3H * CF + G3H * Hdim;
    gru_prep_weights<<<(nconv + 255) / 256, 256, 0, stream>>>(W_ih, W_hh, wih_bf, whh_bf);

    gru_coord_head_kernel<<<Bsz / BM, NT, 0, stream>>>(
        features, last_coord, W_ih, b_ih, b_hh, W_proj, b_proj,
        wih_bf, whh_bf, (float*)d_out);
}
